// POPSSFP4Linear_81003083203670
// MI455X (gfx1250) — compile-verified, matrix operand order checked by reading, not run on hardware
//
#include <hip/hip_runtime.h>

// Reference: x[4,2048,4096] f32, idx[4096,256,16] i32, scales[4096,256] f32,
// bias[4096] f32 -> y[4,2048,4096] f32.   y = x @ W^T + b.
#define MD 8192   // 4*2048 rows of x
#define ND 4096   // OUT_F
#define KD 4096   // IN_F

typedef __attribute__((ext_vector_type(16))) _Float16 v16h;
typedef __attribute__((ext_vector_type(8)))  _Float16 v8h;
typedef __attribute__((ext_vector_type(8)))  float    v8f;

// FP4 E2M1 decode: [0,0.5,1,1.5,2,3,4,6, -0,-0.5,-1,-1.5,-2,-3,-4,-6]
__device__ __forceinline__ float fp4_val(int c) {
    int e = (c >> 1) & 3;
    int m = c & 1;
    float mag = (e == 0) ? 0.5f * (float)m
                         : (float)(1 << (e - 1)) * (1.0f + 0.5f * (float)m);
    return (c & 8) ? -mag : mag;
}

// Load one 16-bit WMMA operand fragment (32 bytes / lane, two b128 loads).
__device__ __forceinline__ void load_frag(v16h& f, const _Float16* p) {
    v8h lo = *(const v8h*)(p);
    v8h hi = *(const v8h*)(p + 16);
#pragma unroll
    for (int t = 0; t < 8; ++t) { f[t] = lo[t]; f[t + 8] = hi[t]; }
}

// ---------------------------------------------------------------------------
// Pass 1: dequantize int32 FP4 codes -> f16 W_dq[ND][KD].
__global__ void __launch_bounds__(256)
fp4_dequant_kernel(const int* __restrict__ widx,
                   const float* __restrict__ wscale,
                   _Float16* __restrict__ Wd) {
    long long t    = (long long)blockIdx.x * blockDim.x + threadIdx.x;
    long long base = t * 8;
    int o = (int)(base >> 12);
    int k = (int)(base & (KD - 1));
    float s = wscale[(o << 8) + (k >> 4)];
    int4 c0 = *(const int4*)(widx + base);
    int4 c1 = *(const int4*)(widx + base + 4);
    v8h out;
    out[0] = (_Float16)(fp4_val(c0.x) * s);
    out[1] = (_Float16)(fp4_val(c0.y) * s);
    out[2] = (_Float16)(fp4_val(c0.z) * s);
    out[3] = (_Float16)(fp4_val(c0.w) * s);
    out[4] = (_Float16)(fp4_val(c1.x) * s);
    out[5] = (_Float16)(fp4_val(c1.y) * s);
    out[6] = (_Float16)(fp4_val(c1.z) * s);
    out[7] = (_Float16)(fp4_val(c1.w) * s);
    *(v8h*)(Wd + base) = out;
}

// ---------------------------------------------------------------------------
// Pass 2: convert x f32 -> f16 once.
__global__ void __launch_bounds__(256)
x_to_f16_kernel(const float* __restrict__ X, _Float16* __restrict__ Xh) {
    long long base = ((long long)blockIdx.x * blockDim.x + threadIdx.x) * 8;
    v8f f = *(const v8f*)(X + base);
    v8h h;
#pragma unroll
    for (int t = 0; t < 8; ++t) h[t] = (_Float16)f[t];
    *(v8h*)(Xh + base) = h;
}

// ---------------------------------------------------------------------------
// Pass 3 (fast): f16 WMMA GEMM with a 2-deep software pipeline.
// Block = 256 threads = 8 waves (2x4); block tile 128(M) x 256(N);
// wave tile 64x64 = 4x4 WMMA tiles; K stepped by 32, pipelined two steps.
__global__ void __launch_bounds__(256)
wmma_gemm_f16(const _Float16* __restrict__ Xh,
              const _Float16* __restrict__ Wd,
              const float* __restrict__ bias,
              float* __restrict__ Y) {
    const int lane = threadIdx.x & 31;
    const int wave = threadIdx.x >> 5;
    const int ln16 = lane & 15;
    const int sel  = lane >> 4;       // which K half-group this lane holds
    const int wr   = wave >> 2;       // 0..1
    const int wc   = wave & 3;        // 0..3
    const int m0   = blockIdx.y * 128 + wr * 64;
    const int n0   = blockIdx.x * 256 + wc * 64;

    v8f acc[4][4];
#pragma unroll
    for (int i = 0; i < 4; ++i)
#pragma unroll
        for (int j = 0; j < 4; ++j)
            acc[i][j] = (v8f){};

    const _Float16* ap[4];
    const _Float16* bp[4];
#pragma unroll
    for (int i = 0; i < 4; ++i)
        ap[i] = Xh + (size_t)(m0 + 16 * i + ln16) * KD + sel * 8;
#pragma unroll
    for (int j = 0; j < 4; ++j)
        bp[j] = Wd + (size_t)(n0 + 16 * j + ln16) * KD + sel * 8;

    // Double-buffered fragments: loads for step k+32 are issued before the
    // WMMA block of step k, so their latency hides under ~16 XDL ops.
    v16h a[2][4], b[2][4];
#pragma unroll
    for (int i = 0; i < 4; ++i) load_frag(a[0][i], ap[i]);
#pragma unroll
    for (int j = 0; j < 4; ++j) load_frag(b[0][j], bp[j]);

#pragma unroll 1
    for (int k0 = 0; k0 < KD; k0 += 64) {
        // Stage A: prefetch k0+32 into buffer 1, compute on buffer 0.
#pragma unroll
        for (int i = 0; i < 4; ++i) load_frag(a[1][i], ap[i] + k0 + 32);
#pragma unroll
        for (int j = 0; j < 4; ++j) load_frag(b[1][j], bp[j] + k0 + 32);
#pragma unroll
        for (int i = 0; i < 4; ++i)
#pragma unroll
            for (int j = 0; j < 4; ++j)
                acc[i][j] = __builtin_amdgcn_wmma_f32_16x16x32_f16(
                    false, a[0][i], false, b[0][j], (short)0, acc[i][j],
                    false, false);

        // Stage B: prefetch k0+64 (wraps to 0 on the last step; 16 harmless
        // extra loads instead of a branch), compute on buffer 1.
        const int k2 = (k0 + 64) & (KD - 1);
#pragma unroll
        for (int i = 0; i < 4; ++i) load_frag(a[0][i], ap[i] + k2);
#pragma unroll
        for (int j = 0; j < 4; ++j) load_frag(b[0][j], bp[j] + k2);
#pragma unroll
        for (int i = 0; i < 4; ++i)
#pragma unroll
            for (int j = 0; j < 4; ++j)
                acc[i][j] = __builtin_amdgcn_wmma_f32_16x16x32_f16(
                    false, a[1][i], false, b[1][j], (short)0, acc[i][j],
                    false, false);
    }

    // Epilogue: C element p maps to M = p + 8*sel, N = ln16.
#pragma unroll
    for (int j = 0; j < 4; ++j) {
        float bj = bias[n0 + 16 * j + ln16];
#pragma unroll
        for (int i = 0; i < 4; ++i) {
            float* yp = Y + (size_t)(m0 + 16 * i + 8 * sel) * ND + n0 + 16 * j + ln16;
#pragma unroll
            for (int p = 0; p < 8; ++p)
                yp[(size_t)p * ND] = acc[i][j][p] + bj;
        }
    }
}

// ---------------------------------------------------------------------------
// Mid tier: GEMM converting x f32->f16 in-loop (only needs 32 MiB ws).
__global__ void __launch_bounds__(256)
wmma_gemm_cvt(const float* __restrict__ X,
              const _Float16* __restrict__ Wd,
              const float* __restrict__ bias,
              float* __restrict__ Y) {
    const int lane = threadIdx.x & 31;
    const int wave = threadIdx.x >> 5;
    const int ln16 = lane & 15;
    const int sel  = lane >> 4;
    const int wr   = wave >> 2;
    const int wc   = wave & 3;
    const int m0   = blockIdx.y * 128 + wr * 64;
    const int n0   = blockIdx.x * 128 + wc * 32;

    v8f acc[4][2];
#pragma unroll
    for (int i = 0; i < 4; ++i)
#pragma unroll
        for (int j = 0; j < 2; ++j) acc[i][j] = (v8f){};

    const float*    ap[4];
    const _Float16* bp[2];
#pragma unroll
    for (int i = 0; i < 4; ++i)
        ap[i] = X + (size_t)(m0 + 16 * i + ln16) * KD + sel * 8;
#pragma unroll
    for (int j = 0; j < 2; ++j)
        bp[j] = Wd + (size_t)(n0 + 16 * j + ln16) * KD + sel * 8;

    for (int k0 = 0; k0 < KD; k0 += 32) {
        v16h a[4], b[2];
#pragma unroll
        for (int i = 0; i < 4; ++i) {
            v8f lo = *(const v8f*)(ap[i] + k0);
            v8f hi = *(const v8f*)(ap[i] + k0 + 16);
#pragma unroll
            for (int t = 0; t < 8; ++t) {
                a[i][t] = (_Float16)lo[t]; a[i][t + 8] = (_Float16)hi[t];
            }
        }
#pragma unroll
        for (int j = 0; j < 2; ++j) load_frag(b[j], bp[j] + k0);
#pragma unroll
        for (int i = 0; i < 4; ++i)
#pragma unroll
            for (int j = 0; j < 2; ++j)
                acc[i][j] = __builtin_amdgcn_wmma_f32_16x16x32_f16(
                    false, a[i], false, b[j], (short)0, acc[i][j], false, false);
    }
#pragma unroll
    for (int j = 0; j < 2; ++j) {
        float bj = bias[n0 + 16 * j + ln16];
#pragma unroll
        for (int i = 0; i < 4; ++i) {
            float* yp = Y + (size_t)(m0 + 16 * i + 8 * sel) * ND + n0 + 16 * j + ln16;
#pragma unroll
            for (int p = 0; p < 8; ++p)
                yp[(size_t)p * ND] = acc[i][j][p] + bj;
        }
    }
}

// ---------------------------------------------------------------------------
// Last-resort fallback (no workspace): naive dot product. Never expected to run.
__global__ void __launch_bounds__(256)
naive_linear_kernel(const float* __restrict__ X, const int* __restrict__ widx,
                    const float* __restrict__ wsc, const float* __restrict__ bias,
                    float* __restrict__ Y) {
    long long t = (long long)blockIdx.x * blockDim.x + threadIdx.x;
    int o = (int)(t % ND);
    int m = (int)(t / ND);
    float s = 0.f;
    for (int k = 0; k < KD; ++k) {
        float w = fp4_val(widx[(size_t)o * KD + k]) * wsc[(o << 8) + (k >> 4)];
        s += X[(size_t)m * KD + k] * w;
    }
    Y[t] = s + bias[o];
}

// ---------------------------------------------------------------------------
extern "C" void kernel_launch(void* const* d_in, const int* in_sizes, int n_in,
                              void* d_out, int out_size, void* d_ws, size_t ws_size,
                              hipStream_t stream) {
    const float* x    = (const float*)d_in[0];
    const int*   widx = (const int*)d_in[1];
    const float* wsc  = (const float*)d_in[2];
    const float* bias = (const float*)d_in[3];
    float*       y    = (float*)d_out;

    const size_t W_BYTES = (size_t)ND * KD * 2;   // 32 MiB
    const size_t X_BYTES = (size_t)MD * KD * 2;   // 64 MiB

    if (ws_size >= W_BYTES + X_BYTES) {
        _Float16* Wd = (_Float16*)d_ws;
        _Float16* Xh = (_Float16*)((char*)d_ws + W_BYTES);
        fp4_dequant_kernel<<<(ND * KD / 8) / 256, 256, 0, stream>>>(widx, wsc, Wd);
        x_to_f16_kernel<<<((size_t)MD * KD / 8) / 256, 256, 0, stream>>>(x, Xh);
        dim3 grid(ND / 256, MD / 128);            // 16 x 64
        wmma_gemm_f16<<<grid, 256, 0, stream>>>(Xh, Wd, bias, y);
    } else if (ws_size >= W_BYTES) {
        _Float16* Wd = (_Float16*)d_ws;
        fp4_dequant_kernel<<<(ND * KD / 8) / 256, 256, 0, stream>>>(widx, wsc, Wd);
        dim3 grid(ND / 128, MD / 128);            // 32 x 64
        wmma_gemm_cvt<<<grid, 256, 0, stream>>>(x, Wd, bias, y);
    } else {
        naive_linear_kernel<<<((size_t)MD * ND) / 256, 256, 0, stream>>>(
            x, widx, wsc, bias, y);
    }
}
